// SimpleGAT_29403346108561
// MI455X (gfx1250) — matrix operand order repaired and checked by reading, not compiled, not run on hardware
//
#include <hip/hip_runtime.h>

// ---------------------------------------------------------------------------
// Types for CDNA5 WMMA (wave32): 16x16x32 bf16 -> f32
// ---------------------------------------------------------------------------
typedef __attribute__((ext_vector_type(16))) __bf16 v16bf;
typedef __attribute__((ext_vector_type(8)))  float  v8f;

#define NEG_SLOPE 0.2f

static __device__ __forceinline__ __bf16 f2bf(float f) {
  // round-to-nearest-even fp32 -> bf16
  unsigned u = __float_as_uint(f);
  unsigned r = (u + 0x7FFFu + ((u >> 16) & 1u)) >> 16;
  unsigned short s = (unsigned short)r;
  return __builtin_bit_cast(__bf16, s);
}

// monotone float<->uint mapping so atomicMax(u32) == float max
static __device__ __forceinline__ unsigned ordf(float f) {
  unsigned u = __float_as_uint(f);
  return (u >> 31) ? ~u : (u | 0x80000000u);
}
static __device__ __forceinline__ float unordf(unsigned u) {
  return (u >> 31) ? __uint_as_float(u & 0x7FFFFFFFu) : __uint_as_float(~u);
}
// ordf(-inf) == 0x007FFFFF  (init value for the max buffers)

// ---------------------------------------------------------------------------
// Pack weight matrix W[K][N] (fp32 row-major) into a contiguous-K B-operand order (A loads match):
// per (ktile,ntile): 32 lanes x 16 bf16.  B 32x16: lanes 0-15 hold K 0..15 of
// column N=lane, lanes 16-31 hold K 16..31; element v within v16bf == K offset.
// ---------------------------------------------------------------------------
__global__ void pack_b_bf16(const float* __restrict__ W, __bf16* __restrict__ bpk,
                            int Kdim, int Ndim) {
  int idx = blockIdx.x * blockDim.x + threadIdx.x;
  int total = (Kdim >> 5) * (Ndim >> 4) * 512;
  if (idx >= total) return;
  int v    = idx & 15;
  int lane = (idx >> 4) & 31;
  int tile = idx >> 9;
  int NT = Ndim >> 4;
  int kt = tile / NT, nt = tile - kt * NT;
  int ncol = nt * 16 + (lane & 15);
  int krow = kt * 32 + ((lane & 16) ? 16 : 0) + v;
  bpk[idx] = f2bf(W[krow * Ndim + ncol]);
}

__global__ void fill_u32_kernel(unsigned* __restrict__ p, unsigned val, int count) {
  int i = blockIdx.x * blockDim.x + threadIdx.x;
  if (i < count) p[i] = val;
}

// ---------------------------------------------------------------------------
// GEMM1: h[n,64] = bf16(x[n,128]) @ bf16(W1[128,64]).  One wave = 16-row tile,
// 4 column tiles, 4 K-steps of 32.  Row clamp (not branch) keeps EXEC full.
// ---------------------------------------------------------------------------
__global__ __launch_bounds__(128) void gemm1_wmma(
    const float* __restrict__ x, const __bf16* __restrict__ bpk,
    float* __restrict__ h, int nrows) {
  const int lane  = threadIdx.x & 31;
  const int mtile = blockIdx.x * 4 + (threadIdx.x >> 5);
  const int row0  = mtile * 16;
  if (row0 >= nrows) return;                 // wave-uniform exit
  int mr = row0 + (lane & 15);
  if (mr >= nrows) mr = nrows - 1;           // clamp loads, mask stores later
  const float* xrow = x + (long)mr * 128;
  const int khalf = (lane & 16) ? 16 : 0;    // contiguous-K lane-half offset (matches B pack)
  const v16bf* bptr = (const v16bf*)bpk;
  v8f acc0 = {}, acc1 = {}, acc2 = {}, acc3 = {};
#pragma unroll
  for (int kt = 0; kt < 4; ++kt) {
    v16bf a;
#pragma unroll
    for (int p = 0; p < 8; ++p) {            // A: K = 0..15 (+16 for hi lanes)
      const int kk = kt * 32 + khalf + 2 * p;
      const float2 f = *(const float2*)(xrow + kk);
      a[2 * p]     = f2bf(f.x);
      a[2 * p + 1] = f2bf(f.y);
    }
    v16bf b0 = bptr[(kt * 4 + 0) * 32 + lane];
    v16bf b1 = bptr[(kt * 4 + 1) * 32 + lane];
    v16bf b2 = bptr[(kt * 4 + 2) * 32 + lane];
    v16bf b3 = bptr[(kt * 4 + 3) * 32 + lane];
    acc0 = __builtin_amdgcn_wmma_f32_16x16x32_bf16(false, a, false, b0, (short)0, acc0, false, false);
    acc1 = __builtin_amdgcn_wmma_f32_16x16x32_bf16(false, a, false, b1, (short)0, acc1, false, false);
    acc2 = __builtin_amdgcn_wmma_f32_16x16x32_bf16(false, a, false, b2, (short)0, acc2, false, false);
    acc3 = __builtin_amdgcn_wmma_f32_16x16x32_bf16(false, a, false, b3, (short)0, acc3, false, false);
  }
  const int mo0 = row0 + ((lane & 16) ? 8 : 0);
  const int nc  = lane & 15;
#pragma unroll
  for (int r = 0; r < 8; ++r) {
    const int mo = mo0 + r;
    if (mo < nrows) {
      float* orow = h + (long)mo * 64 + nc;
      orow[0] = acc0[r]; orow[16] = acc1[r]; orow[32] = acc2[r]; orow[48] = acc3[r];
    }
  }
}

// GEMM2: h2[n,16] = bf16(h1out[n,64]) @ bf16(W2[64,16]).  2 K-steps, 1 col tile.
__global__ __launch_bounds__(128) void gemm2_wmma(
    const float* __restrict__ hin, const __bf16* __restrict__ bpk,
    float* __restrict__ h, int nrows) {
  const int lane  = threadIdx.x & 31;
  const int mtile = blockIdx.x * 4 + (threadIdx.x >> 5);
  const int row0  = mtile * 16;
  if (row0 >= nrows) return;
  int mr = row0 + (lane & 15);
  if (mr >= nrows) mr = nrows - 1;
  const float* arow = hin + (long)mr * 64;
  const int khalf = (lane & 16) ? 16 : 0;
  const v16bf* bptr = (const v16bf*)bpk;
  v8f acc = {};
#pragma unroll
  for (int kt = 0; kt < 2; ++kt) {
    v16bf a;
#pragma unroll
    for (int p = 0; p < 8; ++p) {
      const int kk = kt * 32 + khalf + 2 * p;
      const float2 f = *(const float2*)(arow + kk);
      a[2 * p]     = f2bf(f.x);
      a[2 * p + 1] = f2bf(f.y);
    }
    v16bf b = bptr[kt * 32 + lane];
    acc = __builtin_amdgcn_wmma_f32_16x16x32_bf16(false, a, false, b, (short)0, acc, false, false);
  }
  const int mo0 = row0 + ((lane & 16) ? 8 : 0);
  const int nc  = lane & 15;
#pragma unroll
  for (int r = 0; r < 8; ++r) {
    const int mo = mo0 + r;
    if (mo < nrows) h[(long)mo * 16 + nc] = acc[r];
  }
}

// ---------------------------------------------------------------------------
// Attention dot products
// ---------------------------------------------------------------------------
__global__ void att1_kernel(const float* __restrict__ h, const float* __restrict__ asrc,
                            const float* __restrict__ adst, float* __restrict__ a_s,
                            float* __restrict__ a_d, int n) {
  int idx = blockIdx.x * blockDim.x + threadIdx.x;
  if (idx >= n * 8) return;
  const int node = idx >> 3, hd = idx & 7;
  const float4* hp = (const float4*)(h + (long)node * 64 + hd * 8);
  const float4 h0 = hp[0], h1 = hp[1];
  const float4* sp = (const float4*)(asrc + hd * 8);
  const float4* dp = (const float4*)(adst + hd * 8);
  const float4 s0 = sp[0], s1 = sp[1], d0 = dp[0], d1 = dp[1];
  a_s[idx] = h0.x*s0.x + h0.y*s0.y + h0.z*s0.z + h0.w*s0.w
           + h1.x*s1.x + h1.y*s1.y + h1.z*s1.z + h1.w*s1.w;
  a_d[idx] = h0.x*d0.x + h0.y*d0.y + h0.z*d0.z + h0.w*d0.w
           + h1.x*d1.x + h1.y*d1.y + h1.z*d1.z + h1.w*d1.w;
}

__global__ void att2_kernel(const float* __restrict__ h, const float* __restrict__ asrc,
                            const float* __restrict__ adst, float* __restrict__ a_s,
                            float* __restrict__ a_d, int n) {
  int node = blockIdx.x * blockDim.x + threadIdx.x;
  if (node >= n) return;
  const float* hp = h + (long)node * 16;
  float ds = 0.f, dd = 0.f;
#pragma unroll
  for (int c = 0; c < 16; ++c) { float v = hp[c]; ds += v * asrc[c]; dd += v * adst[c]; }
  a_s[node] = ds; a_d[node] = dd;
}

// ---------------------------------------------------------------------------
// Edge passes (E' = E + N with implicit self-loops appended)
// ---------------------------------------------------------------------------
__global__ void edge_max1_kernel(const int* __restrict__ ei, const float* __restrict__ a_s,
                                 const float* __restrict__ a_d, unsigned* __restrict__ m,
                                 int E, int n) {
  int e = blockIdx.x * blockDim.x + threadIdx.x;
  if (e >= E + n) return;
  const int s = (e < E) ? ei[e]     : (e - E);
  const int d = (e < E) ? ei[E + e] : (e - E);
#pragma unroll
  for (int hd = 0; hd < 8; ++hd) {
    float el = a_s[s * 8 + hd] + a_d[d * 8 + hd];
    el = el > 0.f ? el : NEG_SLOPE * el;
    atomicMax(&m[d * 8 + hd], ordf(el));
  }
}

__global__ void edge_sum1_kernel(const int* __restrict__ ei, const float* __restrict__ a_s,
                                 const float* __restrict__ a_d, const unsigned* __restrict__ m,
                                 const float* __restrict__ h, float* __restrict__ den,
                                 float* __restrict__ outraw, int E, int n) {
  int idx = blockIdx.x * blockDim.x + threadIdx.x;
  if (idx >= (E + n) * 8) return;
  const int e = idx >> 3, hd = idx & 7;
  const int s = (e < E) ? ei[e]     : (e - E);
  const int d = (e < E) ? ei[E + e] : (e - E);
  float el = a_s[s * 8 + hd] + a_d[d * 8 + hd];
  el = el > 0.f ? el : NEG_SLOPE * el;
  const float ex = __expf(el - unordf(m[d * 8 + hd]));
  atomicAdd(&den[d * 8 + hd], ex);
  const float* hs = h + (long)s * 64 + hd * 8;
  float* od = outraw + (long)d * 64 + hd * 8;
#pragma unroll
  for (int o = 0; o < 8; ++o) atomicAdd(&od[o], hs[o] * ex);
}

__global__ void finalize1_kernel(const float* __restrict__ outraw, const float* __restrict__ den,
                                 const float* __restrict__ bias, float* __restrict__ hout, int n) {
  int idx = blockIdx.x * blockDim.x + threadIdx.x;
  if (idx >= n * 64) return;
  const int node = idx >> 6, k = idx & 63, hd = k >> 3;
  float v = outraw[idx] / (den[node * 8 + hd] + 1e-16f) + bias[k];
  hout[idx] = v > 0.f ? v : 0.f;   // conv1 bias then ReLU
}

__global__ void edge_max2_kernel(const int* __restrict__ ei, const float* __restrict__ a_s,
                                 const float* __restrict__ a_d, unsigned* __restrict__ m,
                                 int E, int n) {
  int e = blockIdx.x * blockDim.x + threadIdx.x;
  if (e >= E + n) return;
  const int s = (e < E) ? ei[e]     : (e - E);
  const int d = (e < E) ? ei[E + e] : (e - E);
  float el = a_s[s] + a_d[d];
  el = el > 0.f ? el : NEG_SLOPE * el;
  atomicMax(&m[d], ordf(el));
}

__global__ void edge_sum2_kernel(const int* __restrict__ ei, const float* __restrict__ a_s,
                                 const float* __restrict__ a_d, const unsigned* __restrict__ m,
                                 const float* __restrict__ h, float* __restrict__ den,
                                 float* __restrict__ outraw, int E, int n) {
  int e = blockIdx.x * blockDim.x + threadIdx.x;
  if (e >= E + n) return;
  const int s = (e < E) ? ei[e]     : (e - E);
  const int d = (e < E) ? ei[E + e] : (e - E);
  float el = a_s[s] + a_d[d];
  el = el > 0.f ? el : NEG_SLOPE * el;
  const float ex = __expf(el - unordf(m[d]));
  atomicAdd(&den[d], ex);
  const float* hs = h + (long)s * 16;
  float* od = outraw + (long)d * 16;
#pragma unroll
  for (int c = 0; c < 16; ++c) atomicAdd(&od[c], hs[c] * ex);
}

__global__ void final_kernel(const float* __restrict__ outraw, const float* __restrict__ den,
                             const float* __restrict__ bias, float* __restrict__ out, int n) {
  int node = blockIdx.x * blockDim.x + threadIdx.x;
  if (node >= n) return;
  const float inv = 1.f / (den[node] + 1e-16f);
  float v[16];
  float mx = -3.402823466e38f;
#pragma unroll
  for (int c = 0; c < 16; ++c) {
    v[c] = outraw[(long)node * 16 + c] * inv + bias[c];
    mx = fmaxf(mx, v[c]);
  }
  float s = 0.f;
#pragma unroll
  for (int c = 0; c < 16; ++c) s += __expf(v[c] - mx);
  const float ls = __logf(s);
#pragma unroll
  for (int c = 0; c < 16; ++c) out[(long)node * 16 + c] = v[c] - mx - ls;
}

// ---------------------------------------------------------------------------
extern "C" void kernel_launch(void* const* d_in, const int* in_sizes, int n_in,
                              void* d_out, int out_size, void* d_ws, size_t ws_size,
                              hipStream_t stream) {
  const float* x        = (const float*)d_in[0];
  const int*   ei       = (const int*)  d_in[1];
  const float* W1       = (const float*)d_in[2];
  const float* att_src1 = (const float*)d_in[3];
  const float* att_dst1 = (const float*)d_in[4];
  const float* b1       = (const float*)d_in[5];
  const float* W2       = (const float*)d_in[6];
  const float* att_src2 = (const float*)d_in[7];
  const float* att_dst2 = (const float*)d_in[8];
  const float* b2       = (const float*)d_in[9];
  float* out = (float*)d_out;

  const int n = in_sizes[0] / 128;
  const int E = in_sizes[1] / 2;

  // ---- workspace layout (all offsets 256B aligned) ----
  char* ws = (char*)d_ws;
  size_t off = 0;
  auto alloc = [&](size_t bytes) -> char* {
    char* p = ws + off;
    off += (bytes + 255) & ~(size_t)255;
    return p;
  };
  __bf16* bpk1    = (__bf16*)alloc((size_t)128 * 64 * 2);
  __bf16* bpk2    = (__bf16*)alloc((size_t)64 * 16 * 2);
  float*  h1      = (float*) alloc((size_t)n * 64 * 4);   // also hosts layer-2 block
  float*  h1out   = (float*) alloc((size_t)n * 64 * 4);
  float*  a_s1    = (float*) alloc((size_t)n * 8 * 4);
  float*  a_d1    = (float*) alloc((size_t)n * 8 * 4);
  unsigned* m1    = (unsigned*)alloc((size_t)n * 8 * 4);
  float*  den1    = (float*) alloc((size_t)n * 8 * 4);
  float*  outraw1 = (float*) alloc((size_t)n * 64 * 4);   // contiguous after den1

  // layer-2 buffers alias the h1 region (dead after edge_sum1/finalize1)
  char* l2 = (char*)h1;
  float*    h2      = (float*)l2;
  float*    a_s2    = (float*)(l2 + (size_t)n * 16 * 4);
  float*    a_d2    = a_s2 + n;
  unsigned* m2      = (unsigned*)(a_d2 + n);
  float*    den2    = (float*)(m2 + n);
  float*    outraw2 = den2 + n;                           // contiguous after den2

  const int B = 256;
  auto grid = [](long c, int b) { return (unsigned)((c + b - 1) / b); };

  // 1. pack weights into WMMA B layout (bf16)
  pack_b_bf16<<<grid(8192, B), B, 0, stream>>>(W1, bpk1, 128, 64);
  pack_b_bf16<<<grid(1024, B), B, 0, stream>>>(W2, bpk2, 64, 16);

  // 2. init layer-1 accumulators (ws is poisoned every timing run)
  fill_u32_kernel<<<grid((long)n * 8, B), B, 0, stream>>>(m1, 0x007FFFFFu, n * 8); // ordf(-inf)
  fill_u32_kernel<<<grid((long)n * 72, B), B, 0, stream>>>((unsigned*)den1, 0u, n * 72);

  // 3. layer 1
  const unsigned gtiles = (unsigned)(((n + 15) / 16 + 3) / 4);
  gemm1_wmma<<<gtiles, 128, 0, stream>>>(x, bpk1, h1, n);
  att1_kernel<<<grid((long)n * 8, B), B, 0, stream>>>(h1, att_src1, att_dst1, a_s1, a_d1, n);
  edge_max1_kernel<<<grid((long)E + n, B), B, 0, stream>>>(ei, a_s1, a_d1, m1, E, n);
  edge_sum1_kernel<<<grid(((long)E + n) * 8, B), B, 0, stream>>>(ei, a_s1, a_d1, m1, h1,
                                                                 den1, outraw1, E, n);
  finalize1_kernel<<<grid((long)n * 64, B), B, 0, stream>>>(outraw1, den1, b1, h1out, n);

  // 4. init layer-2 accumulators (h1 now dead; aliased region safe)
  fill_u32_kernel<<<grid((long)n, B), B, 0, stream>>>(m2, 0x007FFFFFu, n);
  fill_u32_kernel<<<grid((long)n * 17, B), B, 0, stream>>>((unsigned*)den2, 0u, n * 17);

  // 5. layer 2
  gemm2_wmma<<<gtiles, 128, 0, stream>>>(h1out, bpk2, h2, n);
  att2_kernel<<<grid((long)n, B), B, 0, stream>>>(h2, att_src2, att_dst2, a_s2, a_d2, n);
  edge_max2_kernel<<<grid((long)E + n, B), B, 0, stream>>>(ei, a_s2, a_d2, m2, E, n);
  edge_sum2_kernel<<<grid((long)E + n, B), B, 0, stream>>>(ei, a_s2, a_d2, m2, h2,
                                                           den2, outraw2, E, n);
  final_kernel<<<grid((long)n, B), B, 0, stream>>>(outraw2, den2, b2, out, n);
}
